// RoPEAttention_52286931861743
// MI455X (gfx1250) — matrix operand order repaired and checked by reading, not compiled, not run on hardware
//
#include <hip/hip_runtime.h>
#include <hip/hip_bf16.h>

// ---------------------------------------------------------------- types
typedef __bf16 bf16;
typedef __attribute__((ext_vector_type(16))) __bf16 v16bf;
typedef __attribute__((ext_vector_type(8)))  float  v8f;
typedef __attribute__((ext_vector_type(4)))  unsigned int u32x4;
typedef __attribute__((ext_vector_type(4)))  float  f32x4;

union Frag {           // one WMMA 16-bit A/B operand: 16 bf16 = 8 VGPRs
  v16bf v;
  u32x4 q[2];
  unsigned u[8];
};

static __device__ __forceinline__ bf16 f2bf(float f) {
  unsigned u = __builtin_bit_cast(unsigned, f);
  unsigned r = (u + 0x7FFFu + ((u >> 16) & 1u)) >> 16;   // round-nearest-even
  unsigned short s = (unsigned short)r;
  return __builtin_bit_cast(bf16, s);
}
static __device__ __forceinline__ float bf2f(bf16 h) {
  unsigned short s = __builtin_bit_cast(unsigned short, h);
  unsigned u = ((unsigned)s) << 16;
  return __builtin_bit_cast(float, u);
}

static __device__ __forceinline__ v8f wmma_bf16(v16bf a, v16bf b, v8f c) {
  return __builtin_amdgcn_wmma_f32_16x16x32_bf16(
      /*neg_a=*/false, a, /*neg_b=*/false, b,
      /*c_mod=*/(short)0, c, /*reuse_a=*/false, /*reuse_b=*/false);
}

// max across lanes at xor-distance {1,2,4,8} within each 16-lane row.
// v_permlane16_b32 (pure VALU) — confirmed lowering on gfx1250.
#if __has_builtin(__builtin_amdgcn_permlane16)
static __device__ __forceinline__ float xormax_step(float v, unsigned s1, unsigned s2) {
  int iv = __builtin_bit_cast(int, v);
  int pv = __builtin_amdgcn_permlane16(iv, iv, (int)s1, (int)s2, false, false);
  return fmaxf(v, __builtin_bit_cast(float, pv));
}
static __device__ __forceinline__ float rowmax16(float v) {
  v = xormax_step(v, 0x67452301u, 0xEFCDAB89u);  // xor 1
  v = xormax_step(v, 0x54761032u, 0xDCFE98BAu);  // xor 2
  v = xormax_step(v, 0x32107654u, 0xBA98FEDCu);  // xor 4
  v = xormax_step(v, 0xFEDCBA98u, 0x76543210u);  // xor 8
  return v;
}
#else
static __device__ __forceinline__ float rowmax16(float v) {
  v = fmaxf(v, __shfl_xor(v, 1, 32));
  v = fmaxf(v, __shfl_xor(v, 2, 32));
  v = fmaxf(v, __shfl_xor(v, 4, 32));
  v = fmaxf(v, __shfl_xor(v, 8, 32));
  return v;
}
#endif

#define D_MODEL 1024
#define N_HEADS 16
#define D_HEAD  64
#define SEQ_L   2048
#define BATCH   2
#define M_ROWS  (BATCH * SEQ_L)   // 4096

// ---------------------------------------------------------------- fp32 -> bf16 (vec4)
__global__ void cvt_f32_bf16_k(const float* __restrict__ in, bf16* __restrict__ out, int n4) {
  int i = blockIdx.x * blockDim.x + threadIdx.x;
  if (i >= n4) return;
  f32x4 v = *(const f32x4*)(in + (size_t)i * 4);
  bf16* o = out + (size_t)i * 4;
  o[0] = f2bf(v.x); o[1] = f2bf(v.y); o[2] = f2bf(v.z); o[3] = f2bf(v.w);
}

// ---------------------------------------------------------------- GEMM: C[M,N] = A[M,K] * W[N,K]^T
// block = 128 threads (4 waves); block tile 64x64, each wave 16 rows x 64 cols.
template <bool BF16OUT>
__global__ __launch_bounds__(128)
void gemm_nt_k(const bf16* __restrict__ A, const bf16* __restrict__ W,
               float* __restrict__ Cf, bf16* __restrict__ Cb,
               int M, int N, int K) {
  const int lane = threadIdx.x & 31;
  const int wave = threadIdx.x >> 5;
  const int lm = lane & 15;        // row (A) / col (B,C) within tile
  const int lh = lane >> 4;        // half-wave selector
  const int m0 = blockIdx.x * 64 + wave * 16;
  const int n0 = blockIdx.y * 64;

  const bf16* arow = A + (size_t)(m0 + lm) * K;
  const bf16* brow[4] = {
    W + (size_t)(n0 + 0  + lm) * K,
    W + (size_t)(n0 + 16 + lm) * K,
    W + (size_t)(n0 + 32 + lm) * K,
    W + (size_t)(n0 + 48 + lm) * K };

  v8f acc[4] = {};
  for (int k0 = 0; k0 < K; k0 += 32) {
    // issue ALL loads for this k-step first, then the 4 WMMAs
    Frag a, b[4];
    a.q[0] = *(const u32x4*)(arow + k0 + lh * 16);
    a.q[1] = *(const u32x4*)(arow + k0 + lh * 16 + 8);
#pragma unroll
    for (int j = 0; j < 4; ++j) {
      b[j].q[0] = *(const u32x4*)(brow[j] + k0 + lh * 16);
      b[j].q[1] = *(const u32x4*)(brow[j] + k0 + lh * 16 + 8);
    }
#pragma unroll
    for (int j = 0; j < 4; ++j)
      acc[j] = wmma_bf16(a.v, b[j].v, acc[j]);
  }
  // C layout: lane holds col lm, rows r + 8*lh (VGPR r)
#pragma unroll
  for (int j = 0; j < 4; ++j) {
#pragma unroll
    for (int r = 0; r < 8; ++r) {
      size_t idx = (size_t)(m0 + r + lh * 8) * N + (n0 + j * 16 + lm);
      float v = acc[j][r];
      if (BF16OUT) Cb[idx] = f2bf(v);
      else         Cf[idx] = v;
    }
  }
}

// ---------------------------------------------------------------- RoPE + relayout (B,L,H*64) -> (B,H,L,64)
__global__ void rope_relayout_k(const bf16* __restrict__ C, bf16* __restrict__ Qh) {
  int idx = blockIdx.x * blockDim.x + threadIdx.x;   // (b,l,h,i) pairs, i in 0..31
  if (idx >= BATCH * SEQ_L * N_HEADS * 32) return;
  int i = idx & 31;
  int h = (idx >> 5) & (N_HEADS - 1);
  int l = (idx >> 9) & (SEQ_L - 1);
  int b = idx >> 20;
  float freq  = __expf(-9.210340371976184f * (float)(2 * i) / (float)D_HEAD);
  float angle = (float)l * freq;
  float s = __sinf(angle), c = __cosf(angle);
  size_t src = ((size_t)(b * SEQ_L + l)) * D_MODEL + h * D_HEAD + 2 * i;
  float xe = bf2f(C[src]), xo = bf2f(C[src + 1]);
  size_t dst = (((size_t)(b * N_HEADS + h)) * SEQ_L + l) * D_HEAD + 2 * i;
  Qh[dst]     = f2bf(xe * c - xo * s);
  Qh[dst + 1] = f2bf(xe * s + xo * c);
}

// ---------------------------------------------------------------- V relayout (B,L,H*64) -> (B,H,64,L)
__global__ void v_relayout_k(const bf16* __restrict__ C, bf16* __restrict__ Vt) {
  int idx = blockIdx.x * blockDim.x + threadIdx.x;
  if (idx >= BATCH * SEQ_L * D_MODEL) return;
  int d = idx & (D_HEAD - 1);
  int h = (idx >> 6) & (N_HEADS - 1);
  int l = (idx >> 10) & (SEQ_L - 1);
  int b = idx >> 21;
  size_t src = ((size_t)(b * SEQ_L + l)) * D_MODEL + h * D_HEAD + d;
  size_t dst = (((size_t)(b * N_HEADS + h)) * D_HEAD + d) * SEQ_L + l;
  Vt[dst] = C[src];
}

// ---------------------------------------------------------------- flash attention (per head)
// grid: (L/64, B*H); block 128 (4 waves); each wave owns 16 query rows.
// Software-pipelined: next K-block and current V-block loads are issued before
// the VALU softmax section so the matrix pipe never waits on L2.
__global__ __launch_bounds__(128)
void attn_k(const bf16* __restrict__ Qh, const bf16* __restrict__ Kh,
            const bf16* __restrict__ Vt, bf16* __restrict__ Obf) {
  __shared__ bf16 plds[4][16 * 32];   // per-wave P staging tile (C-layout -> A-layout)
  const int lane = threadIdx.x & 31;
  const int wave = threadIdx.x >> 5;
  const int lm = lane & 15;
  const int lh = lane >> 4;
  const int bh   = blockIdx.y;                 // b*H + h
  const int qblk = blockIdx.x;
  const int b = bh >> 4, h = bh & (N_HEADS - 1);

  const bf16* Qbase = Qh + (size_t)bh * SEQ_L * D_HEAD;
  const bf16* Kbase = Kh + (size_t)bh * SEQ_L * D_HEAD;
  const bf16* Vbase = Vt + (size_t)bh * D_HEAD * SEQ_L;

  // load Q A-fragments (d split into two k-steps of 32)
  Frag aq[2];
  {
    const bf16* qr = Qbase + (size_t)(qblk * 64 + wave * 16 + lm) * D_HEAD;
    aq[0].q[0] = *(const u32x4*)(qr + lh * 16);
    aq[0].q[1] = *(const u32x4*)(qr + lh * 16 + 8);
    aq[1].q[0] = *(const u32x4*)(qr + 32 + lh * 16);
    aq[1].q[1] = *(const u32x4*)(qr + 32 + lh * 16 + 8);
  }

  // constant all-ones B fragment: rowsum(P) = P x ones via one WMMA
  Frag ones;
#pragma unroll
  for (int j = 0; j < 8; ++j) ones.u[j] = 0x3F803F80u;   // two bf16 1.0

  float mrow[8], lrow[8];
#pragma unroll
  for (int r = 0; r < 8; ++r) { mrow[r] = -1e30f; lrow[r] = 0.0f; }
  v8f acc[4] = {};
  bf16* myp = &plds[wave][0];

  // K-block fragment loader: kf[0,1]=d0..31 x keys {lo,hi}, kf[2,3]=d32..63
  auto load_kblock = [&](Frag* kf, int kb) {
    const bf16* kr0 = Kbase + (size_t)(kb + lm) * D_HEAD + lh * 16;
    const bf16* kr1 = Kbase + (size_t)(kb + 16 + lm) * D_HEAD + lh * 16;
    kf[0].q[0] = *(const u32x4*)(kr0);
    kf[0].q[1] = *(const u32x4*)(kr0 + 8);
    kf[1].q[0] = *(const u32x4*)(kr1);
    kf[1].q[1] = *(const u32x4*)(kr1 + 8);
    kf[2].q[0] = *(const u32x4*)(kr0 + 32);
    kf[2].q[1] = *(const u32x4*)(kr0 + 32 + 8);
    kf[3].q[0] = *(const u32x4*)(kr1 + 32);
    kf[3].q[1] = *(const u32x4*)(kr1 + 32 + 8);
  };

  Frag kf[4];
  load_kblock(kf, 0);

  for (int kb = 0; kb < SEQ_L; kb += 32) {
    // ---- scores S[16q x 32k] from resident K fragments
    v8f s0 = {}, s1 = {};
    s0 = wmma_bf16(aq[0].v, kf[0].v, s0);
    s1 = wmma_bf16(aq[0].v, kf[1].v, s1);
    s0 = wmma_bf16(aq[1].v, kf[2].v, s0);
    s1 = wmma_bf16(aq[1].v, kf[3].v, s1);

    // ---- issue next K-block loads and current V-block loads NOW; the long
    //      VALU softmax below overlaps their latency. Masked address avoids a
    //      branch on the last iteration (EXEC must stay all-ones for WMMA).
    Frag kn[4];
    load_kblock(kn, (kb + 32) & (SEQ_L - 1));
    Frag vf[4];
#pragma unroll
    for (int j = 0; j < 4; ++j) {
      const bf16* vr = Vbase + (size_t)(j * 16 + lm) * SEQ_L + kb + lh * 16;
      vf[j].q[0] = *(const u32x4*)(vr);
      vf[j].q[1] = *(const u32x4*)(vr + 8);
    }

    // ---- online softmax: row-max (VALU permlane), rescale, emit P
#pragma unroll
    for (int r = 0; r < 8; ++r) {
      float a0 = s0[r] * 0.125f, a1 = s1[r] * 0.125f;   // 1/sqrt(64)
      float rv = rowmax16(fmaxf(a0, a1));
      float nm = fmaxf(mrow[r], rv);
      float alpha = __expf(mrow[r] - nm);
      float p0 = __expf(a0 - nm);
      float p1 = __expf(a1 - nm);
      mrow[r] = nm;
      lrow[r] *= alpha;
#pragma unroll
      for (int j = 0; j < 4; ++j) acc[j][r] *= alpha;
      int row = r + lh * 8;
      myp[row * 32 + lm]      = f2bf(p0);   // keys kb..kb+15
      myp[row * 32 + 16 + lm] = f2bf(p1);   // keys kb+16..kb+31
    }
    // ---- read P back as an A 16x32 fragment (LDS ops are in-order per wave)
    Frag pa;
    pa.q[0] = *(const u32x4*)(myp + lm * 32 + lh * 16);
    pa.q[1] = *(const u32x4*)(myp + lm * 32 + lh * 16 + 8);
    // ---- row sums via matrix pipe: rs = P x ones (C-layout matches lrow rows)
    {
      v8f rs = {};
      rs = wmma_bf16(pa.v, ones.v, rs);
#pragma unroll
      for (int r = 0; r < 8; ++r) lrow[r] += rs[r];
    }
    // ---- O += P * V  (V fragments already in flight / resident)
#pragma unroll
    for (int j = 0; j < 4; ++j)
      acc[j] = wmma_bf16(pa.v, vf[j].v, acc[j]);
    // rotate pipeline
#pragma unroll
    for (int t = 0; t < 4; ++t) kf[t] = kn[t];
  }
  // ---- epilogue: normalize & store to (B*L, D_MODEL) bf16 for the O-projection
  float rinv[8];
#pragma unroll
  for (int r = 0; r < 8; ++r) rinv[r] = 1.0f / lrow[r];
#pragma unroll
  for (int j = 0; j < 4; ++j) {
#pragma unroll
    for (int r = 0; r < 8; ++r) {
      int row = qblk * 64 + wave * 16 + r + lh * 8;
      int col = h * D_HEAD + j * 16 + lm;
      float v = acc[j][r] * rinv[r];
      Obf[((size_t)(b * SEQ_L) + row) * D_MODEL + col] = f2bf(v);
    }
  }
}

// ---------------------------------------------------------------- host
extern "C" void kernel_launch(void* const* d_in, const int* in_sizes, int n_in,
                              void* d_out, int out_size, void* d_ws, size_t ws_size,
                              hipStream_t stream) {
  const float* x  = (const float*)d_in[0];
  const float* Wq = (const float*)d_in[1];
  const float* Wk = (const float*)d_in[2];
  const float* Wv = (const float*)d_in[3];
  const float* Wo = (const float*)d_in[4];
  float* out = (float*)d_out;

  char* ws = (char*)d_ws;
  const size_t MB = 1024ull * 1024ull;
  bf16* xbf   = (bf16*)(ws + 0 * MB);    // 8 MB  (4096 x 1024)
  bf16* wqb   = (bf16*)(ws + 8 * MB);    // 2 MB
  bf16* wkb   = (bf16*)(ws + 10 * MB);
  bf16* wvb   = (bf16*)(ws + 12 * MB);
  bf16* wob   = (bf16*)(ws + 14 * MB);
  bf16* cq    = (bf16*)(ws + 16 * MB);   // 8 MB  Q pre-rope (B*L, 1024)
  bf16* ck    = (bf16*)(ws + 24 * MB);   // 8 MB
  bf16* cv    = (bf16*)(ws + 32 * MB);   // 8 MB
  bf16* qh    = (bf16*)(ws + 40 * MB);   // 8 MB  (B,H,L,64)
  bf16* kh    = (bf16*)(ws + 48 * MB);   // 8 MB
  bf16* vt    = (bf16*)(ws + 56 * MB);   // 8 MB  (B,H,64,L)
  bf16* attnb = (bf16*)(ws + 64 * MB);   // 8 MB  (B*L, 1024)

  // 1) bf16 conversion
  {
    int n4x = M_ROWS * D_MODEL / 4;
    cvt_f32_bf16_k<<<(n4x + 255) / 256, 256, 0, stream>>>(x, xbf, n4x);
    int n4w = D_MODEL * D_MODEL / 4;
    cvt_f32_bf16_k<<<(n4w + 255) / 256, 256, 0, stream>>>(Wq, wqb, n4w);
    cvt_f32_bf16_k<<<(n4w + 255) / 256, 256, 0, stream>>>(Wk, wkb, n4w);
    cvt_f32_bf16_k<<<(n4w + 255) / 256, 256, 0, stream>>>(Wv, wvb, n4w);
    cvt_f32_bf16_k<<<(n4w + 255) / 256, 256, 0, stream>>>(Wo, wob, n4w);
  }
  // 2) Q/K/V projections
  dim3 ggrid(M_ROWS / 64, D_MODEL / 64);
  gemm_nt_k<true><<<ggrid, 128, 0, stream>>>(xbf, wqb, nullptr, cq, M_ROWS, D_MODEL, D_MODEL);
  gemm_nt_k<true><<<ggrid, 128, 0, stream>>>(xbf, wkb, nullptr, ck, M_ROWS, D_MODEL, D_MODEL);
  gemm_nt_k<true><<<ggrid, 128, 0, stream>>>(xbf, wvb, nullptr, cv, M_ROWS, D_MODEL, D_MODEL);
  // 3) RoPE + relayout, V transpose
  {
    int n = BATCH * SEQ_L * N_HEADS * 32;
    rope_relayout_k<<<(n + 255) / 256, 256, 0, stream>>>(cq, qh);
    rope_relayout_k<<<(n + 255) / 256, 256, 0, stream>>>(ck, kh);
    int nv = BATCH * SEQ_L * D_MODEL;
    v_relayout_k<<<(nv + 255) / 256, 256, 0, stream>>>(cv, vt);
  }
  // 4) attention
  attn_k<<<dim3(SEQ_L / 64, BATCH * N_HEADS), 128, 0, stream>>>(qh, kh, vt, attnb);
  // 5) output projection (fp32 out)
  gemm_nt_k<false><<<ggrid, 128, 0, stream>>>(attnb, wob, out, nullptr, M_ROWS, D_MODEL, D_MODEL);
}
